// WindowAttentionV2_66468913873037
// MI455X (gfx1250) — matrix operand order repaired and checked by reading, not compiled, not run on hardware
//
#include <hip/hip_runtime.h>

// ---------------------------------------------------------------------------
// Problem constants (SwinV2 window attention)
// ---------------------------------------------------------------------------
#define DIMC   192
#define HEADS  6
#define DH     32
#define NTOK   256      // tokens per window
#define BWIN   256      // windows
#define NW     64       // mask windows
#define CPBH   512
#define NPOS   961      // (2*16-1)^2 relative positions
#define LOG100 4.6051701859880914f

typedef _Float16 v16h  __attribute__((ext_vector_type(16)));
typedef _Float16 f16x8 __attribute__((ext_vector_type(8)));
typedef float    v8f   __attribute__((ext_vector_type(8)));

static __device__ __forceinline__ v8f wmma16(v16h a, v16h b, v8f c) {
  return __builtin_amdgcn_wmma_f32_16x16x32_f16(false, a, false, b, (short)0, c,
                                                false, false);
}

static __device__ __forceinline__ v16h cat8(f16x8 lo, f16x8 hi) {
  return __builtin_shufflevector(lo, hi, 0, 1, 2, 3, 4, 5, 6, 7, 8, 9, 10, 11,
                                 12, 13, 14, 15);
}

// A-matrix fragment (16x32 f16). ISA layout: lanes 0-15 & 16-31 both hold rows
// M=0..15; lanes 0-15 carry K=0..7 / 16..23, lanes 16-31 carry K=8..15 / 24..31.
static __device__ __forceinline__ v16h ldA(const _Float16* base, int rowStride,
                                           int lane) {
  const int hf = lane >> 4;
  const int lr = lane & 15;
  const _Float16* row = base + (size_t)lr * rowStride;
  f16x8 lo = *(const f16x8*)(row + hf * 8);
  f16x8 hi = *(const f16x8*)(row + 16 + hf * 8);
  return cat8(lo, hi);
}

// B-matrix fragment (32x16 f16). Lane l holds column N=l%16; lanes 0-15 carry
// K=0..7 / 16..23, lanes 16-31 carry K=8..15 / 24..31 (same as A).
static __device__ __forceinline__ v16h ldB(const _Float16* base, int rowStride,
                                           int lane, int col) {
  const int hf = lane >> 4;
  const _Float16* p = base + col + (lane & 15);
  v16h r;
#pragma unroll
  for (int s = 0; s < 8; ++s) {
    r[s] = p[(size_t)(hf * 8 + s) * rowStride];
    r[8 + s] = p[(size_t)(16 + hf * 8 + s) * rowStride];
  }
  return r;
}

static __device__ __forceinline__ float redmax16(float v) {
#pragma unroll
  for (int m = 1; m < 16; m <<= 1) v = fmaxf(v, __shfl_xor(v, m, 32));
  return v;
}
static __device__ __forceinline__ float redsum16(float v) {
#pragma unroll
  for (int m = 1; m < 16; m <<= 1) v += __shfl_xor(v, m, 32);
  return v;
}

// ---------------------------------------------------------------------------
// Kernel 1a: CPB MLP  (961 x 2) -> relu(512) -> (961 x 6)
// ---------------------------------------------------------------------------
__global__ void cpb_table_kernel(const float* __restrict__ coords,
                                 const float* __restrict__ w1,
                                 const float* __restrict__ b1,
                                 const float* __restrict__ w2,
                                 float* __restrict__ table) {
  int t = blockIdx.x * blockDim.x + threadIdx.x;
  if (t >= NPOS) return;
  float c0 = coords[t * 2 + 0], c1 = coords[t * 2 + 1];
  float acc[HEADS] = {0.f, 0.f, 0.f, 0.f, 0.f, 0.f};
  for (int j = 0; j < CPBH; ++j) {
    float hv = fmaxf(0.f, w1[j * 2] * c0 + w1[j * 2 + 1] * c1 + b1[j]);
#pragma unroll
    for (int h = 0; h < HEADS; ++h) acc[h] += w2[h * CPBH + j] * hv;
  }
#pragma unroll
  for (int h = 0; h < HEADS; ++h) table[t * HEADS + h] = acc[h];
}

// ---------------------------------------------------------------------------
// Kernel 1b: gather + 16*sigmoid -> bias[h][n][m]  (6,256,256) f32
// ---------------------------------------------------------------------------
__global__ void bias_gather_kernel(const float* __restrict__ table,
                                   const int* __restrict__ rel_index,
                                   float* __restrict__ bias) {
  int tid = blockIdx.x * blockDim.x + threadIdx.x;  // < 6*65536
  int h = tid >> 16;
  int p = tid & 65535;
  float b = table[rel_index[p] * HEADS + h];
  bias[tid] = 16.f / (1.f + __expf(-b));
}

// ---------------------------------------------------------------------------
// Kernel 2: QKV GEMM (65536x576x192) with WMMA, fused q/k row normalization,
// f16 outputs q16/k16/v16 laid out [w][h][n][d].
// Block: 64 rows x 64 cols, 256 threads (8 waves). Wave -> (rowtile, colpair).
// ---------------------------------------------------------------------------
__global__ __launch_bounds__(256) void qkv_kernel(
    const float* __restrict__ x, const float* __restrict__ qkv_w,
    const float* __restrict__ qkv_b, _Float16* __restrict__ q16,
    _Float16* __restrict__ k16, _Float16* __restrict__ v16) {
  __shared__ __align__(16) _Float16 xs[64 * DIMC];   // 24 KB, row-major f16
  __shared__ __align__(16) _Float16 wT[DIMC * 64];   // 24 KB, [k][n] f16

  const int t = threadIdx.x, lane = t & 31, wave = t >> 5;
  const int m0 = blockIdx.x * 64;   // row block (tokens)
  const int n0 = blockIdx.y * 64;   // col block (of 576)

  {  // stage x tile (f32 -> f16)
    int i = t >> 2, qt = t & 3;
    const float* src = x + (size_t)(m0 + i) * DIMC + qt * 48;
    _Float16* dst = xs + i * DIMC + qt * 48;
#pragma unroll
    for (int j = 0; j < 48; ++j) dst[j] = (_Float16)src[j];
  }
  {  // stage W^T tile: wT[k][nl] = qkv_w[n0+nl][k]
    int nl = t & 63, kg = t >> 6;
    const float* src = qkv_w + (size_t)(n0 + nl) * DIMC + kg * 48;
#pragma unroll
    for (int j = 0; j < 48; ++j) wT[(kg * 48 + j) * 64 + nl] = (_Float16)src[j];
  }
  __syncthreads();

  const int rt = wave >> 1;  // 0..3 row tiles of 16
  const int cp = wave & 1;   // 0..1 col chunk of 32 (one head chunk)
  v8f c0 = {0.f, 0.f, 0.f, 0.f, 0.f, 0.f, 0.f, 0.f};
  v8f c1 = c0;
#pragma unroll
  for (int kc = 0; kc < 6; ++kc) {
    v16h a = ldA(xs + rt * 16 * DIMC + kc * 32, DIMC, lane);
    v16h b0 = ldB(wT + kc * 32 * 64, 64, lane, cp * 32);
    v16h b1 = ldB(wT + kc * 32 * 64, 64, lane, cp * 32 + 16);
    c0 = wmma16(a, b0, c0);
    c1 = wmma16(a, b1, c1);
  }

  const int lr = lane & 15, hf = lane >> 4;
  const int colbase = n0 + cp * 32;          // global col of 32-chunk
  const int sec = colbase / DIMC;            // 0=q 1=k 2=v
  const int hh = (colbase % DIMC) / DH;      // head
  const float bq0 = qkv_b[colbase + lr];
  const float bq1 = qkv_b[colbase + 16 + lr];
  _Float16* dst = (sec == 0) ? q16 : (sec == 1) ? k16 : v16;

#pragma unroll
  for (int j = 0; j < 8; ++j) {
    float u0 = c0[j] + bq0;
    float u1 = c1[j] + bq1;
    float ss = redsum16(u0 * u0 + u1 * u1);
    float scl = 1.f;
    if (sec < 2) scl = 1.f / fmaxf(sqrtf(ss), 1e-12f);
    int tok = m0 + rt * 16 + j + hf * 8;
    int w = tok >> 8, n = tok & 255;
    size_t base = (((size_t)(w * HEADS + hh) * NTOK) + n) * DH;
    dst[base + lr] = (_Float16)(u0 * scl);
    dst[base + 16 + lr] = (_Float16)(u1 * scl);
  }
}

// ---------------------------------------------------------------------------
// Kernel 3: per-window attention. Block = one window, 256 threads (8 waves),
// loops over 6 heads. S = qn kn^T via WMMA (K=32=Dh, one op per tile),
// register softmax with lane reductions, P re-staged via LDS to A-layout,
// PV via WMMA, f16 output o16[w][n][h*32+d].
// ---------------------------------------------------------------------------
#define KTS 272  // padded kT row stride (bytes multiple of 16)
__global__ __launch_bounds__(256) void attn_kernel(
    const float* __restrict__ mask, const float* __restrict__ logit_scale,
    const float* __restrict__ bias, const _Float16* __restrict__ q16,
    const _Float16* __restrict__ k16, const _Float16* __restrict__ v16,
    _Float16* __restrict__ o16) {
  __shared__ __align__(16) _Float16 qs[NTOK * DH];   // 16 KB
  __shared__ __align__(16) _Float16 kT[DH * KTS];    // 17 KB (transposed k)
  __shared__ __align__(16) _Float16 vs[NTOK * DH];   // 16 KB
  __shared__ __align__(16) _Float16 pst[8 * 16 * 32];// 8 KB P staging / wave

  const int w = blockIdx.x;
  const int t = threadIdx.x, lane = t & 31, wave = t >> 5;
  const int lr = lane & 15, hf = lane >> 4;
  const float* maskp = mask + (size_t)(w & (NW - 1)) * NTOK * NTOK;

  for (int h = 0; h < HEADS; ++h) {
    const size_t hb = ((size_t)(w * HEADS + h)) * NTOK * DH;
    {  // cooperative loads: q,v row-major; k transposed into kT[d][n]
      const f16x8* gq = (const f16x8*)(q16 + hb + (size_t)t * DH);
      const f16x8* gv = (const f16x8*)(v16 + hb + (size_t)t * DH);
      const f16x8* gk = (const f16x8*)(k16 + hb + (size_t)t * DH);
      f16x8* lq = (f16x8*)(qs + t * DH);
      f16x8* lv = (f16x8*)(vs + t * DH);
      lq[0] = gq[0]; lq[1] = gq[1]; lq[2] = gq[2]; lq[3] = gq[3];
      lv[0] = gv[0]; lv[1] = gv[1]; lv[2] = gv[2]; lv[3] = gv[3];
      f16x8 k0 = gk[0], k1 = gk[1], k2 = gk[2], k3 = gk[3];
#pragma unroll
      for (int d = 0; d < 8; ++d) {
        kT[(d + 0) * KTS + t] = k0[d];
        kT[(d + 8) * KTS + t] = k1[d];
        kT[(d + 16) * KTS + t] = k2[d];
        kT[(d + 24) * KTS + t] = k3[d];
      }
      if (h + 1 < HEADS)  // hint next head's K stream (global_prefetch_b8)
        __builtin_prefetch(k16 + hb + NTOK * DH + (size_t)t * DH, 0, 0);
    }
    __syncthreads();

    const float scale_h = __expf(fminf(logit_scale[h], LOG100));
    const float* biasp = bias + (size_t)h * NTOK * NTOK;

    for (int rti = 0; rti < 2; ++rti) {
      const int rt = wave + rti * 8;  // row tile 0..15
      v16h aq = ldA(qs + rt * 16 * DH, DH, lane);

      v8f c[16];
#pragma unroll
      for (int nt = 0; nt < 16; ++nt) {
        v8f z = {0.f, 0.f, 0.f, 0.f, 0.f, 0.f, 0.f, 0.f};
        c[nt] = wmma16(aq, ldB(kT, KTS, lane, nt * 16), z);
      }

      // scale + bias + mask, then row softmax (rows striped across lanes 0-15
      // or 16-31; reduce with 16-lane xor shuffles)
      float rmax[8];
#pragma unroll
      for (int j = 0; j < 8; ++j) rmax[j] = -3.0e38f;
#pragma unroll
      for (int nt = 0; nt < 16; ++nt) {
#pragma unroll
        for (int j = 0; j < 8; ++j) {
          int M = rt * 16 + j + hf * 8;
          int N = nt * 16 + lr;
          float v = c[nt][j] * scale_h + biasp[M * NTOK + N] + maskp[M * NTOK + N];
          c[nt][j] = v;
          rmax[j] = fmaxf(rmax[j], v);
        }
      }
#pragma unroll
      for (int j = 0; j < 8; ++j) rmax[j] = redmax16(rmax[j]);
      float rsum[8];
#pragma unroll
      for (int j = 0; j < 8; ++j) rsum[j] = 0.f;
#pragma unroll
      for (int nt = 0; nt < 16; ++nt) {
#pragma unroll
        for (int j = 0; j < 8; ++j) {
          float p = __expf(c[nt][j] - rmax[j]);
          c[nt][j] = p;
          rsum[j] += p;
        }
      }
      float rinv[8];
#pragma unroll
      for (int j = 0; j < 8; ++j) rinv[j] = 1.f / redsum16(rsum[j]);

      // P @ V : K=256 in 8 chunks of 32; P re-laid C->A via per-wave LDS
      v8f co0 = {0.f, 0.f, 0.f, 0.f, 0.f, 0.f, 0.f, 0.f};
      v8f co1 = co0;
      _Float16* pb = pst + wave * (16 * 32);
#pragma unroll
      for (int kc = 0; kc < 8; ++kc) {
#pragma unroll
        for (int lc = 0; lc < 2; ++lc) {
          const int nt = kc * 2 + lc;
#pragma unroll
          for (int j = 0; j < 8; ++j)
            pb[(j + hf * 8) * 32 + lc * 16 + lr] =
                (_Float16)(c[nt][j] * rinv[j]);
        }
        asm volatile("s_wait_dscnt 0" ::: "memory");  // same-wave LDS RAW
        v16h pa = ldA(pb, 32, lane);
        v16h bv0 = ldB(vs + kc * 32 * DH, DH, lane, 0);
        v16h bv1 = ldB(vs + kc * 32 * DH, DH, lane, 16);
        co0 = wmma16(pa, bv0, co0);
        co1 = wmma16(pa, bv1, co1);
      }

#pragma unroll
      for (int j = 0; j < 8; ++j) {
        int n = rt * 16 + j + hf * 8;
        size_t ob = ((size_t)w * NTOK + n) * DIMC + h * DH;
        o16[ob + lr] = (_Float16)co0[j];
        o16[ob + 16 + lr] = (_Float16)co1[j];
      }
    }
    __syncthreads();
  }
}

// ---------------------------------------------------------------------------
// Kernel 4: output projection (65536x192x192) with WMMA, fp32 output + bias.
// Block: 64 rows x 64 cols, 256 threads (8 waves).
// ---------------------------------------------------------------------------
__global__ __launch_bounds__(256) void proj_kernel(
    const _Float16* __restrict__ o16, const float* __restrict__ proj_w,
    const float* __restrict__ proj_b, float* __restrict__ out) {
  __shared__ __align__(16) _Float16 os[64 * DIMC];
  __shared__ __align__(16) _Float16 pwT[DIMC * 64];

  const int t = threadIdx.x, lane = t & 31, wave = t >> 5;
  const int m0 = blockIdx.x * 64;
  const int n0 = blockIdx.y * 64;

  {
    int i = t >> 2, qt = t & 3;
    const _Float16* src = o16 + (size_t)(m0 + i) * DIMC + qt * 48;
    _Float16* dst = os + i * DIMC + qt * 48;
#pragma unroll
    for (int j = 0; j < 48; ++j) dst[j] = src[j];
  }
  {
    int nl = t & 63, kg = t >> 6;
    const float* src = proj_w + (size_t)(n0 + nl) * DIMC + kg * 48;
#pragma unroll
    for (int j = 0; j < 48; ++j)
      pwT[(kg * 48 + j) * 64 + nl] = (_Float16)src[j];
  }
  __syncthreads();

  const int rt = wave >> 1;
  const int cp = wave & 1;
  v8f c0 = {0.f, 0.f, 0.f, 0.f, 0.f, 0.f, 0.f, 0.f};
  v8f c1 = c0;
#pragma unroll
  for (int kc = 0; kc < 6; ++kc) {
    v16h a = ldA(os + rt * 16 * DIMC + kc * 32, DIMC, lane);
    v16h b0 = ldB(pwT + kc * 32 * 64, 64, lane, cp * 32);
    v16h b1 = ldB(pwT + kc * 32 * 64, 64, lane, cp * 32 + 16);
    c0 = wmma16(a, b0, c0);
    c1 = wmma16(a, b1, c1);
  }

  const int lr = lane & 15, hf = lane >> 4;
  const int col0 = n0 + cp * 32 + lr;
  const int col1 = col0 + 16;
  const float pb0 = proj_b[col0];
  const float pb1 = proj_b[col1];
#pragma unroll
  for (int j = 0; j < 8; ++j) {
    int tok = m0 + rt * 16 + j + hf * 8;
    out[(size_t)tok * DIMC + col0] = c0[j] + pb0;
    out[(size_t)tok * DIMC + col1] = c1[j] + pb1;
  }
}

// ---------------------------------------------------------------------------
// Launch
// ---------------------------------------------------------------------------
extern "C" void kernel_launch(void* const* d_in, const int* in_sizes, int n_in,
                              void* d_out, int out_size, void* d_ws,
                              size_t ws_size, hipStream_t stream) {
  const float* x = (const float*)d_in[0];
  const float* mask = (const float*)d_in[1];
  const float* qkv_w = (const float*)d_in[2];
  const float* qkv_b = (const float*)d_in[3];
  const float* proj_w = (const float*)d_in[4];
  const float* proj_b = (const float*)d_in[5];
  const float* logit_scale = (const float*)d_in[6];
  const float* cpb_w1 = (const float*)d_in[7];
  const float* cpb_b1 = (const float*)d_in[8];
  const float* cpb_w2 = (const float*)d_in[9];
  const float* coords = (const float*)d_in[10];
  const int* rel_index = (const int*)d_in[11];
  float* out = (float*)d_out;

  char* wsb = (char*)d_ws;
  float* bias = (float*)(wsb + 0);                       // 6*65536*4 = 1.5 MB
  float* table = (float*)(wsb + 0x180000);               // 961*6*4
  _Float16* q16 = (_Float16*)(wsb + 0x186000);
  const size_t QE = (size_t)BWIN * HEADS * NTOK * DH;    // 12.58M elems
  _Float16* k16 = q16 + QE;
  _Float16* v16 = k16 + QE;
  _Float16* o16 = v16 + QE;

  cpb_table_kernel<<<4, 256, 0, stream>>>(coords, cpb_w1, cpb_b1, cpb_w2,
                                          table);
  bias_gather_kernel<<<(HEADS * 65536) / 256, 256, 0, stream>>>(table,
                                                                rel_index,
                                                                bias);
  qkv_kernel<<<dim3(1024, 9), 256, 0, stream>>>(x, qkv_w, qkv_b, q16, k16,
                                                v16);
  attn_kernel<<<BWIN, 256, 0, stream>>>(mask, logit_scale, bias, q16, k16,
                                        v16, o16);
  proj_kernel<<<dim3(1024, 3), 256, 0, stream>>>(o16, proj_w, proj_b, out);
}
